// BGI_70901320122525
// MI455X (gfx1250) — compile-verified, matrix operand order checked by reading, not run on hardware
//
#include <hip/hip_runtime.h>
#include <hip/hip_bf16.h>
#include <math.h>

// ---------------------------------------------------------------------------
// Sizes (compile-time constants from the reference)
// ---------------------------------------------------------------------------
#define BB   2048
#define T9   9
#define NN   62
#define CC   10
#define TT   8
#define DD   62
#define ED   3
#define KK   2
#define PP   100
#define HH   62
#define BN   (BB * NN)          // 126976 GRU rows
#define PSTR 64                 // padded gate stride (62 -> 64)
#define NCOL 192                // 3 gates * 64

typedef __attribute__((ext_vector_type(16))) _Float16 v16h;
typedef __attribute__((ext_vector_type(8)))  _Float16 v8h;
typedef __attribute__((ext_vector_type(8)))  float    v8f;

// Fast nonlinearities built on the v_rcp_f32 / v_exp_f32 trans ops (avoids the
// IEEE div expansion: div_scale + rcp + Newton FMAs per gate element).
__device__ __forceinline__ float sigmoidf_(float x) {
    return __builtin_amdgcn_rcpf(1.0f + __expf(-x));
}
__device__ __forceinline__ float tanhf_(float x) {
    // 1 - 2/(exp(2x)+1); saturates correctly: rcp(inf)=0 -> +/-1
    return 1.0f - 2.0f * __builtin_amdgcn_rcpf(1.0f + __expf(2.0f * x));
}

// Load a 16-bit WMMA fragment (A-style lane layout) from an LDS row of 64 halfs.
// A operand: rowbase = tile + (lane&15)*64 (lane's matrix row).
// B operand: rowbase = Bmat + col*64      (lane's matrix column, col = 16*j + (lane&15)).
// ISA 16-bit A 16x32 layout: lanes 0-15 hold K {0..7,16..23}(+32*kc), lanes 16-31 hold K {8..15,24..31}(+32*kc).
__device__ __forceinline__ v16h frag_ld(const _Float16* rowbase, int lane, int kc) {
    int hi = (lane >> 4) & 1;
    const _Float16* p = rowbase + kc * 32 + hi * 8;
    v8h lo = *(const v8h*)(p);
    v8h hv = *(const v8h*)(p + 16);
    v16h r;
#pragma unroll
    for (int i = 0; i < 8; ++i) { r[i] = lo[i]; r[i + 8] = hv[i]; }
    return r;
}

__device__ __forceinline__ v8f wmma_f16(v16h a, v16h b, v8f c) {
    // D = A(16x32 f16) * B(32x16 f16) + C(16x16 f32)
    return __builtin_amdgcn_wmma_f32_16x16x32_f16(
        /*neg_a=*/false, a, /*neg_b=*/false, b,
        /*c_mod=*/(short)0, c, /*reuse_a=*/false, /*reuse_b=*/false);
}

// ---------------------------------------------------------------------------
// K0: adjacency softmax A = softmax(relu(E E^T)) and temporal softmax alpha
// ---------------------------------------------------------------------------
__global__ void k0_precompute(const float* __restrict__ E,
                              const float* __restrict__ Tparam,
                              float* __restrict__ wsA,
                              float* __restrict__ wsAlpha) {
    int tid = threadIdx.x;
    if (tid < NN) {
        float logit[NN];
        float mx = -1e30f;
        float e0 = E[tid * ED + 0], e1 = E[tid * ED + 1], e2 = E[tid * ED + 2];
        for (int j = 0; j < NN; ++j) {
            float d = e0 * E[j * ED + 0] + e1 * E[j * ED + 1] + e2 * E[j * ED + 2];
            d = fmaxf(d, 0.0f);
            logit[j] = d;
            mx = fmaxf(mx, d);
        }
        float s = 0.0f;
        for (int j = 0; j < NN; ++j) { float e = __expf(logit[j] - mx); logit[j] = e; s += e; }
        float inv = 1.0f / s;
        for (int j = 0; j < NN; ++j) wsA[tid * NN + j] = logit[j] * inv;
    } else if (tid >= 64 && tid < 64 + TT) {
        int t = tid - 64;                        // window length t+1
        float mx = -1e30f;
        for (int s = 0; s <= t; ++s) mx = fmaxf(mx, Tparam[t * T9 + s]);
        float ex[T9]; float sum = 0.0f;
        for (int s = 0; s < T9; ++s) {
            float e = (s <= t) ? __expf(Tparam[t * T9 + s] - mx) : 0.0f;
            ex[s] = e; sum += e;
        }
        float inv = 1.0f / sum;
        for (int s = 0; s < T9; ++s) wsAlpha[t * T9 + s] = ex[s] * inv;
    }
}

// ---------------------------------------------------------------------------
// K1: per-node weights  w[t,n,k,c,o] = sum_e E[n,e] Wp[t,e,k,c,o]
//     per-node bias     b[t,n,o]     = sum_e E[n,e] bp[t,e,o]
// ---------------------------------------------------------------------------
#define W_ELEMS (TT * NN * KK * CC * DD)   // 615040
#define B_ELEMS (TT * NN * DD)             // 30752
__global__ void k1_weights(const float* __restrict__ E,
                           const float* __restrict__ Wp,
                           const float* __restrict__ bp,
                           float* __restrict__ wsW,
                           float* __restrict__ wsB) {
    int idx = blockIdx.x * blockDim.x + threadIdx.x;
    if (idx < W_ELEMS) {
        int o = idx % DD;
        int c = (idx / DD) % CC;
        int k = (idx / (DD * CC)) % KK;
        int n = (idx / (DD * CC * KK)) % NN;
        int t = idx / (DD * CC * KK * NN);
        float acc = 0.0f;
        for (int e = 0; e < ED; ++e)
            acc += E[n * ED + e] * Wp[((((t * ED + e) * KK + k) * CC + c) * DD) + o];
        wsW[idx] = acc;
    } else if (idx < W_ELEMS + B_ELEMS) {
        int i2 = idx - W_ELEMS;
        int o = i2 % DD;
        int n = (i2 / DD) % NN;
        int t = i2 / (DD * NN);
        float acc = 0.0f;
        for (int e = 0; e < ED; ++e)
            acc += E[n * ED + e] * bp[(t * ED + e) * DD + o];
        wsB[i2] = acc;
    }
}

// ---------------------------------------------------------------------------
// K2: zz[b,t] = sigmoid( sum_p PI[b,p] * Wz[t,p] )   (82 MB streaming read)
// ---------------------------------------------------------------------------
__global__ void k2_zz(const float* __restrict__ PI,
                      const float* __restrict__ Wz,
                      float* __restrict__ wsZZ) {
    __shared__ float red[256 * TT];
    int b = blockIdx.x, tid = threadIdx.x;
    float acc[TT];
#pragma unroll
    for (int t = 0; t < TT; ++t) acc[t] = 0.0f;
    const float* pib = PI + (size_t)b * (PP * PP);
    for (int p = tid; p < PP * PP; p += 256) {
        float v = pib[p];
#pragma unroll
        for (int t = 0; t < TT; ++t) acc[t] += v * Wz[t * (PP * PP) + p];
    }
#pragma unroll
    for (int t = 0; t < TT; ++t) red[tid * TT + t] = acc[t];
    __syncthreads();
    for (int s2 = 128; s2 > 0; s2 >>= 1) {
        if (tid < s2) {
#pragma unroll
            for (int t = 0; t < TT; ++t) red[tid * TT + t] += red[(tid + s2) * TT + t];
        }
        __syncthreads();
    }
    if (tid < TT) wsZZ[b * TT + tid] = sigmoidf_(red[tid]);
}

// ---------------------------------------------------------------------------
// K3: fused TLSGCN + GRU.  2 waves / block, each wave owns 16 GRU rows.
//     Per step: build xt (62-dim GRU input) on the fly -> LDS fp16 tile ->
//     WMMA gi/gh with padded gate stride 64 -> gate math on C-layout regs.
// ---------------------------------------------------------------------------
#define WPB 2   // waves per block
__global__ __launch_bounds__(32 * WPB, 1)
void k3_gru(const float* __restrict__ x,
            const float* __restrict__ wsA,
            const float* __restrict__ wsAlpha,
            const float* __restrict__ wsW,
            const float* __restrict__ wsB,
            const float* __restrict__ wsZZ,
            const float* __restrict__ W_ih,
            const float* __restrict__ W_hh,
            const float* __restrict__ b_ih,
            const float* __restrict__ b_hh,
            float* __restrict__ rowsum) {
    __shared__ _Float16 Bih[NCOL * PSTR];              // 24 KB  W_ih^T padded, col-major for B
    __shared__ _Float16 Bhh[NCOL * PSTR];              // 24 KB
    __shared__ float    bihp[NCOL], bhhp[NCOL];
    __shared__ _Float16 xtile[WPB][16 * PSTR];         // fp16 A-tiles
    __shared__ _Float16 htile[WPB][16 * PSTR];
    __shared__ float    uv[WPB][16][2][CC];
    __shared__ float    rs[WPB][16];

    const int tid  = threadIdx.x;
    const int wid  = tid >> 5;
    const int lane = tid & 31;
    const int half = lane >> 4;
    const int l15  = lane & 15;
    const int nthr = 32 * WPB;

    // ---- one-time block init: padded fp16 weights + biases into LDS ----
    for (int idx = tid; idx < NCOL * PSTR; idx += nthr) {
        int col = idx >> 6, k = idx & 63;
        int gate = col >> 6, o = col & 63;
        float wih = 0.0f, whh = 0.0f;
        if (o < DD && k < DD) {
            int g = gate * HH + o;                     // row of W_*h [3H, D]
            wih = W_ih[g * DD + k];
            whh = W_hh[g * HH + k];
        }
        Bih[idx] = (_Float16)wih;
        Bhh[idx] = (_Float16)whh;
    }
    for (int idx = tid; idx < NCOL; idx += nthr) {
        int gate = idx >> 6, o = idx & 63;
        bihp[idx] = (o < DD) ? b_ih[gate * HH + o] : 0.0f;
        bhhp[idx] = (o < DD) ? b_hh[gate * HH + o] : 0.0f;
    }
    for (int idx = tid; idx < WPB * 16 * PSTR; idx += nthr)
        ((_Float16*)htile)[idx] = (_Float16)0.0f;
    __syncthreads();

    const int r0w = (blockIdx.x * WPB + wid) * 16;     // wave's first GRU row
    const int r   = r0w + l15;                         // this lane-pair's row
    const int bb  = r / NN;
    const int nn  = r % NN;
    const float* xb   = x + (size_t)bb * (T9 * NN * CC);
    const float* Arow = wsA + nn * NN;

    // hidden state in C-layout registers: hst[j][i] = h[row i+half*8][col 16j+l15]
    v8f hst[4];
#pragma unroll
    for (int j = 0; j < 4; ++j)
#pragma unroll
        for (int i = 0; i < 8; ++i) hst[j][i] = 0.0f;

    for (int t = 0; t < TT; ++t) {
        const float zzv   = wsZZ[bb * TT + t];
        const float* alph = wsAlpha + t * T9;

        // ---- phase 1: per-row reduced vectors u[c], v[c] (lane pair splits c) --
        {
            int c0 = half * 5;
#pragma unroll
            for (int cc = 0; cc < 5; ++cc) {
                int c = c0 + cc;
                const float* xf = xb + (t + 1) * (NN * CC) + c;   // x[b][t+1][.][c]
                float xval = xf[nn * CC];
                float xA = 0.0f;
                for (int m2 = 0; m2 < NN; ++m2) xA += Arow[m2] * xf[m2 * CC];
                float xw = 0.0f;
#pragma unroll
                for (int s = 0; s < T9; ++s) xw += alph[s] * xb[s * (NN * CC) + nn * CC + c];
                uv[wid][l15][0][c] = zzv * xval + xw;   // pairs with w[t,n,0,:,:]
                uv[wid][l15][1][c] = zzv * xA;          // pairs with w[t,n,1,:,:]
            }
        }
        __syncthreads();

        // ---- phase 2: xt[o] = u.w0[:,o] + v.w1[:,o] + zz*b[o]  -> fp16 LDS ----
        {
            const float* w0 = wsW + (((size_t)t * NN + nn) * KK + 0) * (CC * DD);
            const float* w1 = wsW + (((size_t)t * NN + nn) * KK + 1) * (CC * DD);
            const float* bv = wsB + ((size_t)t * NN + nn) * DD;
            _Float16* xt = xtile[wid];
            int o0 = half * 31;
            for (int oo = 0; oo < 31; ++oo) {
                int o = o0 + oo;
                float acc = zzv * bv[o];
#pragma unroll
                for (int c = 0; c < CC; ++c)
                    acc += uv[wid][l15][0][c] * w0[c * DD + o] +
                           uv[wid][l15][1][c] * w1[c * DD + o];
                if (!(acc == acc) || fabsf(acc) > 3.0e38f) acc = 0.0f;  // nan/inf -> 0
                xt[l15 * PSTR + o] = (_Float16)acc;
            }
            xt[l15 * PSTR + 62 + half] = (_Float16)0.0f;               // K padding
        }
        __syncthreads();

        // ---- phase 3: WMMA gi/gh + gate math on C-layout registers ----------
        {
            v16h ax0 = frag_ld(xtile[wid] + l15 * PSTR, lane, 0);
            v16h ax1 = frag_ld(xtile[wid] + l15 * PSTR, lane, 1);
            v16h ah0 = frag_ld(htile[wid] + l15 * PSTR, lane, 0);
            v16h ah1 = frag_ld(htile[wid] + l15 * PSTR, lane, 1);
#pragma unroll
            for (int j = 0; j < 4; ++j) {
                v8f air, aiz, ain, ahr, ahz, ahn;
#pragma unroll
                for (int i = 0; i < 8; ++i) {
                    air[i] = 0.0f; aiz[i] = 0.0f; ain[i] = 0.0f;
                    ahr[i] = 0.0f; ahz[i] = 0.0f; ahn[i] = 0.0f;
                }
#pragma unroll
                for (int kc = 0; kc < 2; ++kc) {
                    v16h ax = kc ? ax1 : ax0;
                    v16h ah = kc ? ah1 : ah0;
                    air = wmma_f16(ax, frag_ld(Bih + (16 * (j + 0) + l15) * PSTR, lane, kc), air);
                    aiz = wmma_f16(ax, frag_ld(Bih + (16 * (j + 4) + l15) * PSTR, lane, kc), aiz);
                    ain = wmma_f16(ax, frag_ld(Bih + (16 * (j + 8) + l15) * PSTR, lane, kc), ain);
                    ahr = wmma_f16(ah, frag_ld(Bhh + (16 * (j + 0) + l15) * PSTR, lane, kc), ahr);
                    ahz = wmma_f16(ah, frag_ld(Bhh + (16 * (j + 4) + l15) * PSTR, lane, kc), ahz);
                    ahn = wmma_f16(ah, frag_ld(Bhh + (16 * (j + 8) + l15) * PSTR, lane, kc), ahn);
                }
                float bir = bihp[16 * j + l15];
                float biz = bihp[64 + 16 * j + l15];
                float bin_ = bihp[128 + 16 * j + l15];
                float bhr = bhhp[16 * j + l15];
                float bhz = bhhp[64 + 16 * j + l15];
                float bhn = bhhp[128 + 16 * j + l15];
#pragma unroll
                for (int i = 0; i < 8; ++i) {
                    float gr = sigmoidf_(air[i] + bir + ahr[i] + bhr);
                    float gz = sigmoidf_(aiz[i] + biz + ahz[i] + bhz);
                    float gn = tanhf_(ain[i] + bin_ + gr * (ahn[i] + bhn));
                    hst[j][i] = (1.0f - gz) * gn + gz * hst[j][i];
                }
            }
            // write new hidden state back as the next step's fp16 A-tile
#pragma unroll
            for (int j = 0; j < 4; ++j)
#pragma unroll
                for (int i = 0; i < 8; ++i)
                    htile[wid][(i + half * 8) * PSTR + 16 * j + l15] = (_Float16)hst[j][i];
        }
        __syncthreads();
    }

    // ---- epilogue: rowsum[r] = sum_o h[r][o] (padded cols are exactly 0) ----
    if (lane < 16) rs[wid][lane] = 0.0f;
    __syncthreads();
#pragma unroll
    for (int i = 0; i < 8; ++i) {
        float v = hst[0][i] + hst[1][i] + hst[2][i] + hst[3][i];
        atomicAdd(&rs[wid][i + half * 8], v);
    }
    __syncthreads();
    if (lane < 16) rowsum[r0w + lane] = rs[wid][lane];
}

// ---------------------------------------------------------------------------
// K4: out[b] = relu( relu(rowsum[b,:] W1^T + b1) W2^T + b2 )
// ---------------------------------------------------------------------------
__global__ void k4_fc(const float* __restrict__ rowsum,
                      const float* __restrict__ W1, const float* __restrict__ b1,
                      const float* __restrict__ W2, const float* __restrict__ b2,
                      float* __restrict__ out) {
    __shared__ float s[NN];
    __shared__ float hid[256];
    int b = blockIdx.x, tid = threadIdx.x;
    if (tid < NN) s[tid] = rowsum[b * NN + tid];
    __syncthreads();
    float acc = b1[tid];
    for (int n = 0; n < NN; ++n) acc += s[n] * W1[tid * NN + n];
    hid[tid] = fmaxf(acc, 0.0f);
    __syncthreads();
    if (tid < 3) {
        float a = b2[tid];
        for (int j = 0; j < 256; ++j) a += hid[j] * W2[tid * 256 + j];
        out[b * 3 + tid] = fmaxf(a, 0.0f);
    }
}

// ---------------------------------------------------------------------------
// launch
// ---------------------------------------------------------------------------
extern "C" void kernel_launch(void* const* d_in, const int* in_sizes, int n_in,
                              void* d_out, int out_size, void* d_ws, size_t ws_size,
                              hipStream_t stream) {
    const float* x      = (const float*)d_in[0];
    const float* PI     = (const float*)d_in[1];
    const float* E      = (const float*)d_in[2];
    const float* Wp     = (const float*)d_in[3];
    const float* bp     = (const float*)d_in[4];
    const float* Tparam = (const float*)d_in[5];
    const float* Wz     = (const float*)d_in[6];
    const float* W_ih   = (const float*)d_in[7];
    const float* W_hh   = (const float*)d_in[8];
    const float* b_ih   = (const float*)d_in[9];
    const float* b_hh   = (const float*)d_in[10];
    const float* W1     = (const float*)d_in[11];
    const float* b1     = (const float*)d_in[12];
    const float* W2     = (const float*)d_in[13];
    const float* b2     = (const float*)d_in[14];
    float* out = (float*)d_out;
    float* ws  = (float*)d_ws;

    float* wsA     = ws;                         // 3844
    float* wsAlpha = wsA + 3844;                 // 72
    float* wsW     = wsAlpha + 72;               // 615040
    float* wsB     = wsW + W_ELEMS;              // 30752
    float* wsZZ    = wsB + B_ELEMS;              // 16384
    float* wsRS    = wsZZ + BB * TT;             // 126976

    k0_precompute<<<1, 256, 0, stream>>>(E, Tparam, wsA, wsAlpha);
    k1_weights<<<(W_ELEMS + B_ELEMS + 255) / 256, 256, 0, stream>>>(E, Wp, bp, wsW, wsB);
    k2_zz<<<BB, 256, 0, stream>>>(PI, Wz, wsZZ);
    k3_gru<<<BN / (16 * WPB), 32 * WPB, 0, stream>>>(x, wsA, wsAlpha, wsW, wsB, wsZZ,
                                                     W_ih, W_hh, b_ih, b_hh, wsRS);
    k4_fc<<<BB, 256, 0, stream>>>(wsRS, W1, b1, W2, b2, out);
}
